// GNN_13597866459204
// MI455X (gfx1250) — matrix operand order repaired and checked by reading, not compiled, not run on hardware
//
#include <hip/hip_runtime.h>

// ---------------------------------------------------------------------------
// GNN message passing on MI455X (gfx1250): bf16 WMMA GEMMs, f32 accumulate.
// Hot GEMM loops are explicitly double-buffered (two named tile structs,
// k-loop unrolled x2) so loads for tile k+32 stay in flight across the WMMAs
// of tile k -- avoids the s_wait_loadcnt 0x0 serialization the scheduler
// produced when left to its own occupancy heuristics.
// ---------------------------------------------------------------------------

#define HIDDEN 1024
#define MSG    1024
#define NB     256
#define NCAT   80
#define ROWS   (NB * NCAT)      // 20480
#define PADL   511              // (HIDDEN-1)/2

typedef __attribute__((ext_vector_type(16))) __bf16 v16bf;
typedef __attribute__((ext_vector_type(8)))  float  v8f;

union BF16x16 { v16bf v; uint4 q[2]; unsigned int u[8]; unsigned short s[16]; };

__device__ __forceinline__ unsigned short f2bf(float f) {
  union { float f; unsigned int u; } v; v.f = f;
  unsigned int r = v.u + 0x7FFFu + ((v.u >> 16) & 1u);   // round-to-nearest-even
  return (unsigned short)(r >> 16);
}
__device__ __forceinline__ float bf2f(unsigned short u) {
  union { unsigned int u; float f; } v; v.u = ((unsigned int)u) << 16; return v.f;
}

__device__ __forceinline__ v8f bf_mma(const BF16x16& A, const BF16x16& B, v8f c) {
  return __builtin_amdgcn_wmma_f32_16x16x32_bf16(false, A.v, false, B.v,
                                                 (short)0, c, false, false);
}

// ---------------- f32 -> bf16 weight conversion ----------------------------
__global__ __launch_bounds__(256) void k_f2bf(const float* __restrict__ src,
                                              unsigned short* __restrict__ dst,
                                              int n) {
  int i = blockIdx.x * 256 + threadIdx.x;
  if (i < n) dst[i] = f2bf(src[i]);
}

// ---------------- hidden_0 init: x embedded at column PADL -----------------
__global__ __launch_bounds__(256) void k_init(const float* __restrict__ x,
                                              float* __restrict__ hf32,
                                              unsigned short* __restrict__ hbf) {
  int i = blockIdx.x * 256 + threadIdx.x;   // ROWS*HIDDEN = 20,971,520
  int col = i & (HIDDEN - 1);
  int row = i >> 10;
  float v = (col == PADL) ? x[row] : 0.0f;
  hf32[i] = v;
  hbf[i]  = f2bf(v);
}

// ======================= GEMM1: m = ReLU(h @ W1^T + b1) ====================
struct G1T { BF16x16 A, B0, B1, B2, B3; };

__device__ __forceinline__ void g1_load(G1T& T, const unsigned short* ap,
                                        const unsigned short* b0, const unsigned short* b1,
                                        const unsigned short* b2, const unsigned short* b3,
                                        int k0) {
  T.A.q[0]  = *(const uint4*)(ap + k0);
  T.A.q[1]  = *(const uint4*)(ap + k0 + 8);
  T.B0.q[0] = *(const uint4*)(b0 + k0); T.B0.q[1] = *(const uint4*)(b0 + k0 + 8);
  T.B1.q[0] = *(const uint4*)(b1 + k0); T.B1.q[1] = *(const uint4*)(b1 + k0 + 8);
  T.B2.q[0] = *(const uint4*)(b2 + k0); T.B2.q[1] = *(const uint4*)(b2 + k0 + 8);
  T.B3.q[0] = *(const uint4*)(b3 + k0); T.B3.q[1] = *(const uint4*)(b3 + k0 + 8);
}

__device__ __forceinline__ void g1_mma(v8f acc[4], const G1T& T) {
  acc[0] = bf_mma(T.A, T.B0, acc[0]);
  acc[1] = bf_mma(T.A, T.B1, acc[1]);
  acc[2] = bf_mma(T.A, T.B2, acc[2]);
  acc[3] = bf_mma(T.A, T.B3, acc[3]);
}

// one wave -> 16x64 output (4 accumulators share one A tile)
__global__ __launch_bounds__(256) void k_gemm1(
    const unsigned short* __restrict__ hbf,    // [ROWS, HIDDEN] bf16
    const unsigned short* __restrict__ W1b,    // [MSG, HIDDEN]  bf16
    const float* __restrict__ b1,              // [MSG]
    unsigned short* __restrict__ m_bf)         // [ROWS, MSG]    bf16
{
  const int lane = threadIdx.x & 31;
  const int wave = threadIdx.x >> 5;
  const int tile = blockIdx.x * 8 + wave;      // (ROWS/16)*(MSG/64) = 20480
  const int rbase = (tile >> 4) * 16;
  const int cg    = (tile & 15) * 64;
  const int hs = lane >> 4;                    // lane half select
  const int arow = rbase + (lane & 15);
  const int c0   = cg + (lane & 15);

  const unsigned short* ap = hbf + (size_t)arow * HIDDEN + hs * 16;
  const unsigned short* b0 = W1b + (size_t)(c0     ) * HIDDEN + hs * 16;
  const unsigned short* b1p = W1b + (size_t)(c0 + 16) * HIDDEN + hs * 16;
  const unsigned short* b2 = W1b + (size_t)(c0 + 32) * HIDDEN + hs * 16;
  const unsigned short* b3 = W1b + (size_t)(c0 + 48) * HIDDEN + hs * 16;

  v8f acc[4] = {{}, {}, {}, {}};
  G1T Ta, Tb;
  g1_load(Ta, ap, b0, b1p, b2, b3, 0);
  for (int k0 = 0; k0 < HIDDEN - 64; k0 += 64) {
    g1_load(Tb, ap, b0, b1p, b2, b3, k0 + 32);
    g1_mma(acc, Ta);
    g1_load(Ta, ap, b0, b1p, b2, b3, k0 + 64);
    g1_mma(acc, Tb);
  }
  g1_load(Tb, ap, b0, b1p, b2, b3, HIDDEN - 32);
  g1_mma(acc, Ta);
  g1_mma(acc, Tb);

  const int rout = rbase + hs * 8;
#pragma unroll
  for (int t = 0; t < 4; ++t) {
    const int c = c0 + 16 * t;
    const float bias = b1[c];
#pragma unroll
    for (int v = 0; v < 8; ++v) {
      float val = acc[t][v] + bias;
      val = val > 0.0f ? val : 0.0f;
      m_bf[(size_t)(rout + v) * MSG + c] = f2bf(val);
    }
  }
}

// ------- per-batch sum over N, then overwrite m with msg = (sum-m)/(N-1) ----
// each thread owns one (b, c) column: safe in-place transform
__global__ __launch_bounds__(256) void k_msgsum(unsigned short* __restrict__ m_bf) {
  const int b = blockIdx.x >> 2;                           // 256 batches
  const int c = ((blockIdx.x & 3) << 8) + threadIdx.x;     // 0..1023
  unsigned short* p = m_bf + (size_t)b * NCAT * MSG + c;
  float s = 0.0f;
#pragma unroll 8
  for (int n = 0; n < NCAT; ++n) s += bf2f(p[(size_t)n * MSG]);
  const float inv = 1.0f / (float)(NCAT - 1);
#pragma unroll 8
  for (int n = 0; n < NCAT; ++n) {
    const size_t o = (size_t)n * MSG;
    p[o] = f2bf((s - bf2f(p[o])) * inv);
  }
}

// ============== fused GRU: 6 gate GEMMs + gating, new hidden ===============
struct GRT { BF16x16 Am, Ah, B0, B1, B2, B3, B4, B5; };

__device__ __forceinline__ void gr_load(GRT& T,
    const unsigned short* mp, const unsigned short* ap,
    const unsigned short* bir, const unsigned short* biz, const unsigned short* bin,
    const unsigned short* bhr, const unsigned short* bhz, const unsigned short* bhn,
    int k0) {
  T.Am.q[0] = *(const uint4*)(mp + k0);  T.Am.q[1] = *(const uint4*)(mp + k0 + 8);
  T.Ah.q[0] = *(const uint4*)(ap + k0);  T.Ah.q[1] = *(const uint4*)(ap + k0 + 8);
  T.B0.q[0] = *(const uint4*)(bir + k0); T.B0.q[1] = *(const uint4*)(bir + k0 + 8);
  T.B1.q[0] = *(const uint4*)(biz + k0); T.B1.q[1] = *(const uint4*)(biz + k0 + 8);
  T.B2.q[0] = *(const uint4*)(bin + k0); T.B2.q[1] = *(const uint4*)(bin + k0 + 8);
  T.B3.q[0] = *(const uint4*)(bhr + k0); T.B3.q[1] = *(const uint4*)(bhr + k0 + 8);
  T.B4.q[0] = *(const uint4*)(bhz + k0); T.B4.q[1] = *(const uint4*)(bhz + k0 + 8);
  T.B5.q[0] = *(const uint4*)(bhn + k0); T.B5.q[1] = *(const uint4*)(bhn + k0 + 8);
}

__device__ __forceinline__ void gr_mma(v8f acc[6], const GRT& T) {
  acc[0] = bf_mma(T.Am, T.B0, acc[0]);
  acc[1] = bf_mma(T.Am, T.B1, acc[1]);
  acc[2] = bf_mma(T.Am, T.B2, acc[2]);
  acc[3] = bf_mma(T.Ah, T.B3, acc[3]);
  acc[4] = bf_mma(T.Ah, T.B4, acc[4]);
  acc[5] = bf_mma(T.Ah, T.B5, acc[5]);
}

__global__ __launch_bounds__(256) void k_gru(
    const unsigned short* __restrict__ msg_bf,  // [ROWS, MSG]  (transformed m)
    const unsigned short* __restrict__ hbf_cur, // [ROWS, HIDDEN]
    const unsigned short* __restrict__ Wihb,    // [3H, MSG]
    const unsigned short* __restrict__ Whhb,    // [3H, HIDDEN]
    const float* __restrict__ bih,              // [3H]
    const float* __restrict__ bhh,              // [3H]
    float* __restrict__ hf32,                   // in-place f32 state
    unsigned short* __restrict__ hbf_next)      // bf16 shadow for next round
{
  const int lane = threadIdx.x & 31;
  const int wave = threadIdx.x >> 5;
  const int tile = blockIdx.x * 8 + wave;       // 81920 tiles over [ROWS x H]
  const int rbase = (tile >> 6) * 16;
  const int cbase = (tile & 63) * 16;
  const int hs = lane >> 4;
  const int arow = rbase + (lane & 15);
  const int bcol = cbase + (lane & 15);

  const unsigned short* mp = msg_bf  + (size_t)arow * MSG    + hs * 16;
  const unsigned short* ap = hbf_cur + (size_t)arow * HIDDEN + hs * 16;

  const unsigned short* bir = Wihb + (size_t)(0 * HIDDEN + bcol) * MSG    + hs * 16;
  const unsigned short* biz = Wihb + (size_t)(1 * HIDDEN + bcol) * MSG    + hs * 16;
  const unsigned short* bin = Wihb + (size_t)(2 * HIDDEN + bcol) * MSG    + hs * 16;
  const unsigned short* bhr = Whhb + (size_t)(0 * HIDDEN + bcol) * HIDDEN + hs * 16;
  const unsigned short* bhz = Whhb + (size_t)(1 * HIDDEN + bcol) * HIDDEN + hs * 16;
  const unsigned short* bhn = Whhb + (size_t)(2 * HIDDEN + bcol) * HIDDEN + hs * 16;

  v8f acc[6] = {{}, {}, {}, {}, {}, {}};
  GRT Ta, Tb;
  gr_load(Ta, mp, ap, bir, biz, bin, bhr, bhz, bhn, 0);
  for (int k0 = 0; k0 < HIDDEN - 64; k0 += 64) {
    gr_load(Tb, mp, ap, bir, biz, bin, bhr, bhz, bhn, k0 + 32);
    gr_mma(acc, Ta);
    gr_load(Ta, mp, ap, bir, biz, bin, bhr, bhz, bhn, k0 + 64);
    gr_mma(acc, Tb);
  }
  gr_load(Tb, mp, ap, bir, biz, bin, bhr, bhz, bhn, HIDDEN - 32);
  gr_mma(acc, Ta);
  gr_mma(acc, Tb);

  const int rout = rbase + hs * 8;
  const float c_ir = bih[bcol], c_iz = bih[HIDDEN + bcol], c_in = bih[2 * HIDDEN + bcol];
  const float c_hr = bhh[bcol], c_hz = bhh[HIDDEN + bcol], c_hn = bhh[2 * HIDDEN + bcol];
#pragma unroll
  for (int v = 0; v < 8; ++v) {
    const size_t idx = (size_t)(rout + v) * HIDDEN + bcol;
    float ir = acc[0][v] + c_ir, iz = acc[1][v] + c_iz, in_ = acc[2][v] + c_in;
    float hr = acc[3][v] + c_hr, hz = acc[4][v] + c_hz, hn  = acc[5][v] + c_hn;
    float rg = 1.0f / (1.0f + __expf(-(ir + hr)));
    float zg = 1.0f / (1.0f + __expf(-(iz + hz)));
    float ng = tanhf(in_ + rg * hn);
    float ho = hf32[idx];
    float hv = (1.0f - zg) * ng + zg * ho;
    hf32[idx]     = hv;
    hbf_next[idx] = f2bf(hv);
  }
}

// ---------------- readout: out[b,n] = hidden . w_s + b_s -------------------
__global__ __launch_bounds__(256) void k_out(const float* __restrict__ hf32,
                                             const float* __restrict__ w_s,
                                             const float* __restrict__ b_s,
                                             float* __restrict__ out) {
  const int lane = threadIdx.x & 31;
  const int wave = threadIdx.x >> 5;
  const int row  = blockIdx.x * 8 + wave;       // ROWS rows, 2560 blocks
  const float* p = hf32 + (size_t)row * HIDDEN;
  float s = 0.0f;
#pragma unroll 4
  for (int j = lane; j < HIDDEN; j += 32) s += p[j] * w_s[j];
#pragma unroll
  for (int off = 16; off > 0; off >>= 1) s += __shfl_xor(s, off, 32);
  if (lane == 0) out[row] = s + b_s[0];
}

// ---------------------------------------------------------------------------
extern "C" void kernel_launch(void* const* d_in, const int* in_sizes, int n_in,
                              void* d_out, int out_size, void* d_ws, size_t ws_size,
                              hipStream_t stream) {
  (void)in_sizes; (void)n_in; (void)out_size; (void)ws_size;
  const float* x   = (const float*)d_in[0];
  const float* W1  = (const float*)d_in[1];
  const float* b1  = (const float*)d_in[2];
  const float* Wih = (const float*)d_in[3];
  const float* Whh = (const float*)d_in[4];
  const float* bih = (const float*)d_in[5];
  const float* bhh = (const float*)d_in[6];
  const float* w_s = (const float*)d_in[7];
  const float* b_s = (const float*)d_in[8];
  float* out = (float*)d_out;

  char* p = (char*)d_ws;
  float* hf32 = (float*)p;                 p += (size_t)ROWS * HIDDEN * 4;
  unsigned short* hbf0 = (unsigned short*)p; p += (size_t)ROWS * HIDDEN * 2;
  unsigned short* hbf1 = (unsigned short*)p; p += (size_t)ROWS * HIDDEN * 2;
  unsigned short* m_bf = (unsigned short*)p; p += (size_t)ROWS * MSG * 2;
  unsigned short* W1b  = (unsigned short*)p; p += (size_t)MSG * HIDDEN * 2;
  unsigned short* Wihb = (unsigned short*)p; p += (size_t)3 * HIDDEN * MSG * 2;
  unsigned short* Whhb = (unsigned short*)p; p += (size_t)3 * HIDDEN * HIDDEN * 2;

  k_f2bf<<<4096, 256, 0, stream>>>(W1, W1b, MSG * HIDDEN);
  k_f2bf<<<12288, 256, 0, stream>>>(Wih, Wihb, 3 * HIDDEN * MSG);
  k_f2bf<<<12288, 256, 0, stream>>>(Whh, Whhb, 3 * HIDDEN * HIDDEN);
  k_init<<<(ROWS * HIDDEN) / 256, 256, 0, stream>>>(x, hf32, hbf0);

  unsigned short* cur = hbf0;
  unsigned short* nxt = hbf1;
  for (int t = 0; t < 3; ++t) {
    k_gemm1<<<2560, 256, 0, stream>>>(cur, W1b, b1, m_bf);
    k_msgsum<<<1024, 256, 0, stream>>>(m_bf);
    k_gru<<<10240, 256, 0, stream>>>(m_bf, cur, Wihb, Whhb, bih, bhh,
                                     hf32, nxt);
    unsigned short* t2 = cur; cur = nxt; nxt = t2;
  }
  k_out<<<2560, 256, 0, stream>>>(hf32, w_s, b_s, out);
}
